// OuterProductMean_26242250179311
// MI455X (gfx1250) — matrix operand order repaired and checked by reading, not compiled, not run on hardware
//
#include <hip/hip_runtime.h>

// OuterProductMean: IN_C=256, MID_C=32, OUT_C=128, B=1, M=128, N=256
#define IN_C  256
#define MID_C 32
#define OUT_C 128
#define M_DIM 128
#define N_DIM 256

typedef __attribute__((ext_vector_type(16))) __bf16        v16bf;
typedef __attribute__((ext_vector_type(8)))  float         v8f;
typedef __attribute__((ext_vector_type(4)))  unsigned int  u32x4;

union FragU { u32x4 q[2]; v16bf v; };

__device__ __forceinline__ unsigned short f2bf(float f) {
  unsigned int u = __float_as_uint(f);
  u += 0x7FFFu + ((u >> 16) & 1u);     // round-to-nearest-even
  return (unsigned short)(u >> 16);
}

// ---------------------------------------------------------------------------
// Kernel 1: LayerNorm + left/right projections, emit bf16 in WMMA-friendly
// layouts:  abuf[(i*32+x)*128 + m]  (A rows = (i,x), K = m)
//           bbuf[(j*32+y)*128 + m]  (B cols = (j,y), K = m; column-major)
// One wave per (m,n) row; lane = output channel.
// ---------------------------------------------------------------------------
__global__ __launch_bounds__(256) void ln_proj_kernel(
    const float* __restrict__ x1d, const float* __restrict__ mask,
    const float* __restrict__ nw,  const float* __restrict__ nb,
    const float* __restrict__ wl,  const float* __restrict__ bl,
    const float* __restrict__ wr,  const float* __restrict__ br,
    unsigned short* __restrict__ abuf, unsigned short* __restrict__ bbuf)
{
  __shared__ float xn_s[8][IN_C];
  const int wave = threadIdx.x >> 5;
  const int lane = threadIdx.x & 31;
  const int row  = blockIdx.x * 8 + wave;      // 0..32767
  const int m = row >> 8;                      // 0..127
  const int n = row & 255;                     // 0..255
  const float* xr = x1d + (size_t)row * IN_C;

  float x[8];
  float s = 0.f, s2 = 0.f;
  #pragma unroll
  for (int ii = 0; ii < 8; ++ii) {
    float v = xr[lane + 32 * ii];
    x[ii] = v; s += v; s2 += v * v;
  }
  #pragma unroll
  for (int off = 16; off >= 1; off >>= 1) {
    s  += __shfl_xor(s,  off, 32);
    s2 += __shfl_xor(s2, off, 32);
  }
  const float mu  = s  * (1.0f / IN_C);
  const float var = s2 * (1.0f / IN_C) - mu * mu;
  const float inv = rsqrtf(var + 1e-5f);
  #pragma unroll
  for (int ii = 0; ii < 8; ++ii) {
    int k = lane + 32 * ii;
    xn_s[wave][k] = (x[ii] - mu) * inv * nw[k] + nb[k];
  }
  __syncthreads();

  float accL = 0.f, accR = 0.f;
  #pragma unroll 8
  for (int k = 0; k < IN_C; ++k) {
    float xk = xn_s[wave][k];
    accL += xk * wl[k * MID_C + lane];
    accR += xk * wr[k * MID_C + lane];
  }
  const float mv = mask[row];
  accL = (accL + bl[lane]) * mv;
  accR = (accR + br[lane]) * mv;
  abuf[((size_t)n * 32 + lane) * 128 + m] = f2bf(accL);
  bbuf[((size_t)n * 32 + lane) * 128 + m] = f2bf(accR);
}

// ---------------------------------------------------------------------------
// Kernel 2: prep — wf transposed to bf16 [o][xy], and norm[i,j]=Σ_m mask·mask
// ---------------------------------------------------------------------------
__global__ __launch_bounds__(256) void prep_kernel(
    const float* __restrict__ wf, const float* __restrict__ mask,
    unsigned short* __restrict__ wfb, float* __restrict__ nrm)
{
  const int t = blockIdx.x * 256 + threadIdx.x;
  if (t < OUT_C * 1024) {
    int o = t >> 10, xy = t & 1023;
    wfb[t] = f2bf(wf[(size_t)xy * OUT_C + o]);
  }
  if (t < N_DIM * N_DIM) {
    int i = t >> 8, j = t & 255;
    float acc = 0.f;
    for (int mm = 0; mm < M_DIM; ++mm)
      acc += mask[mm * N_DIM + i] * mask[mm * N_DIM + j];
    nrm[t] = acc;
  }
}

// ---------------------------------------------------------------------------
// Kernel 3: fused outer-product GEMM + wf projection.
// Block = 4 i's x 4 j's.  Stage A: C1[128x128] = A(128xK=128) * B, 8 waves,
// each wave owns a 16-row strip (32 wmma).  C1 -> LDS bf16 [pair][xy] (pad 8).
// Stage B: out[16 pairs x 128] with K=1024 (32 wmma per wave).
// ---------------------------------------------------------------------------
__global__ __launch_bounds__(256) void opm_kernel(
    const unsigned short* __restrict__ abuf,
    const unsigned short* __restrict__ bbuf,
    const unsigned short* __restrict__ wfb,
    const float* __restrict__ nrm,
    const float* __restrict__ bfv,
    float* __restrict__ out)
{
  __shared__ unsigned short x2d_s[16 * 1032];   // 33,024 bytes
  const int lane = threadIdx.x & 31;
  const int w    = threadIdx.x >> 5;
  const int kh   = (lane < 16) ? 0 : 1;         // K-half select per WMMA layout
  const int l15  = lane & 15;
  const int i_base = blockIdx.y * 4;
  const int j_base = blockIdx.x * 4;

  // ---------------- Stage A ----------------
  v8f acc[8] = {};
  const unsigned short* aptr =
      abuf + ((size_t)(i_base * 32 + w * 16 + l15)) * 128;

  #pragma unroll
  for (int kt = 0; kt < 4; ++kt) {
    const int k0 = kt * 32;
    FragU af;
    af.q[0] = *reinterpret_cast<const u32x4*>(aptr + k0 + kh * 16);
    af.q[1] = *reinterpret_cast<const u32x4*>(aptr + k0 + kh * 16 + 8);
    const int mb = k0 + kh * 16;
    #pragma unroll
    for (int ct = 0; ct < 8; ++ct) {
      const unsigned short* bptr =
          bbuf + ((size_t)(j_base * 32 + ct * 16 + l15)) * 128 + mb;
      FragU bfr;
      bfr.q[0] = *reinterpret_cast<const u32x4*>(bptr);
      bfr.q[1] = *reinterpret_cast<const u32x4*>(bptr + 8);
      acc[ct] = __builtin_amdgcn_wmma_f32_16x16x32_bf16(
          false, af.v, false, bfr.v, (short)0, acc[ct], false, false);
    }
  }

  // ---------------- C1 -> LDS (bf16, [pair p][xy], row stride 1032) --------
  #pragma unroll
  for (int ct = 0; ct < 8; ++ct) {
    const int col = ct * 16 + l15;              // 0..127
    const int jl = col >> 5, y = col & 31;
    #pragma unroll
    for (int v = 0; v < 8; ++v) {
      const int r  = w * 16 + v + kh * 8;       // stage-A row 0..127
      const int il = r >> 5, xx = r & 31;
      x2d_s[(il * 4 + jl) * 1032 + xx * 32 + y] = f2bf(acc[ct][v]);
    }
  }
  __syncthreads();

  // ---------------- Stage B ----------------
  v8f oacc = {};
  const int oc = w * 16 + l15;                       // output channel
  const unsigned short* wrow  = wfb + (size_t)oc * 1024;
  const unsigned short* arow2 = x2d_s + l15 * 1032;  // pair row = lane%16
  for (int kt = 0; kt < 32; ++kt) {
    const int k0 = kt * 32;
    FragU af;
    af.q[0] = *reinterpret_cast<const u32x4*>(arow2 + k0 + kh * 16);
    af.q[1] = *reinterpret_cast<const u32x4*>(arow2 + k0 + kh * 16 + 8);
    FragU bfr;
    const unsigned short* bp = wrow + k0 + kh * 16;
    bfr.q[0] = *reinterpret_cast<const u32x4*>(bp);
    bfr.q[1] = *reinterpret_cast<const u32x4*>(bp + 8);
    oacc = __builtin_amdgcn_wmma_f32_16x16x32_bf16(
        false, af.v, false, bfr.v, (short)0, oacc, false, false);
  }

  const float bias = bfv[oc];
  #pragma unroll
  for (int v = 0; v < 8; ++v) {
    const int p = v + kh * 8;                  // pair index (D-matrix row)
    const int i = i_base + (p >> 2);
    const int j = j_base + (p & 3);
    const float nv = nrm[i * N_DIM + j] + 0.001f;
    out[((size_t)i * N_DIM + j) * OUT_C + oc] = (oacc[v] + bias) / nv;
  }
}

// ---------------------------------------------------------------------------
extern "C" void kernel_launch(void* const* d_in, const int* in_sizes, int n_in,
                              void* d_out, int out_size, void* d_ws, size_t ws_size,
                              hipStream_t stream) {
  const float* x1d  = (const float*)d_in[0];
  const float* mask = (const float*)d_in[1];
  const float* nw   = (const float*)d_in[2];
  const float* nb   = (const float*)d_in[3];
  const float* wl   = (const float*)d_in[4];
  const float* bl   = (const float*)d_in[5];
  const float* wr   = (const float*)d_in[6];
  const float* br   = (const float*)d_in[7];
  const float* wf   = (const float*)d_in[8];
  const float* bfv  = (const float*)d_in[9];
  float* out = (float*)d_out;

  char* ws = (char*)d_ws;
  unsigned short* abuf = (unsigned short*)(ws);                          // 2 MB
  unsigned short* bbuf = (unsigned short*)(ws + (2u << 20));             // 2 MB
  unsigned short* wfb  = (unsigned short*)(ws + (4u << 20));             // 256 KB
  float*          nrm  = (float*)(ws + (4u << 20) + (256u << 10));       // 256 KB

  ln_proj_kernel<<<4096, 256, 0, stream>>>(x1d, mask, nw, nb, wl, bl, wr, br,
                                           abuf, bbuf);
  prep_kernel<<<512, 256, 0, stream>>>(wf, mask, wfb, nrm);
  dim3 g2(64, 64);
  opm_kernel<<<g2, 256, 0, stream>>>(abuf, bbuf, wfb, nrm, bfv, out);
}
